// LinearAttention_86371792322778
// MI455X (gfx1250) — matrix operand order repaired and checked by reading, not compiled, not run on hardware
//
#include <hip/hip_runtime.h>
#include <hip/hip_bf16.h>

typedef __attribute__((ext_vector_type(16))) _Float16 v16h;
typedef __attribute__((ext_vector_type(8)))  _Float16 v8h;
typedef __attribute__((ext_vector_type(8)))  float    v8f;

#define WMMA_F32_16x16x32_F16(A, B, C) \
  __builtin_amdgcn_wmma_f32_16x16x32_f16(false, (A), false, (B), (short)0, (C), false, false)

#if __has_builtin(__builtin_amdgcn_sched_barrier)
#define LA_SCHED_FENCE() __builtin_amdgcn_sched_barrier(0)
#else
#define LA_SCHED_FENCE()
#endif

static constexpr int Bv = 8;
static constexpr int Nv = 4096;
static constexpr int Cv = 512;
static constexpr int MT = Bv * Nv;   // 32768 rows total
static constexpr int KSTEPS = Cv / 32;

// A fragment (16x32 f16): lane lo holds K {0..15}, hi {16..31}.
// p must already include the row base, K base and +sel*16.
__device__ __forceinline__ v16h la_load_a(const _Float16* p) {
    const v8h lo = *(const v8h*)(p);
    const v8h hi = *(const v8h*)(p + 8);
    v16h a;
#pragma unroll
    for (int i = 0; i < 8; ++i) { a[i] = lo[i]; a[i + 8] = hi[i]; }
    return a;
}

// ---------------------------------------------------------------------------
// Streaming f32 -> f16 conversion, 8 elements per thread
// ---------------------------------------------------------------------------
__global__ __launch_bounds__(256)
void la_cvt_f16_kernel(const float* __restrict__ src, _Float16* __restrict__ dst,
                       int n8) {
    const int i = blockIdx.x * blockDim.x + threadIdx.x;
    if (i >= n8) return;
    const float4 f0 = *(const float4*)(src + (size_t)i * 8);
    const float4 f1 = *(const float4*)(src + (size_t)i * 8 + 4);
    v8h h;
    h[0] = (_Float16)f0.x; h[1] = (_Float16)f0.y;
    h[2] = (_Float16)f0.z; h[3] = (_Float16)f0.w;
    h[4] = (_Float16)f1.x; h[5] = (_Float16)f1.y;
    h[6] = (_Float16)f1.z; h[7] = (_Float16)f1.w;
    *(v8h*)(dst + (size_t)i * 8) = h;
}

__global__ void la_zero_kernel(float* __restrict__ p, int n) {
    int i = blockIdx.x * blockDim.x + threadIdx.x;
    if (i < n) p[i] = 0.0f;
}

// ---------------------------------------------------------------------------
// GEMM1: qk[m,d] = sum_c x[m,c] * W[d,c]   (M=32768, N=1024, K=512), f16 in.
// Wave: 16x64 tile; fully unrolled software pipeline with a sched_barrier
// pinning each step's prefetch ahead of the previous step's WMMAs.
// Epilogue: ELU+1 -> q_h / k_h (f16); k columns atomically summed -> ksum.
// ---------------------------------------------------------------------------
__global__ __launch_bounds__(256)
void la_qkproj_kernel(const _Float16* __restrict__ xh, const _Float16* __restrict__ Wh,
                      _Float16* __restrict__ qh, _Float16* __restrict__ kh,
                      float* __restrict__ ksum) {
    const int bid  = blockIdx.x;        // 0..4095
    const int mt   = bid >> 1;
    const int half = bid & 1;           // 0 -> q columns, 1 -> k columns
    const int w    = threadIdx.x >> 5;
    const int lane = threadIdx.x & 31;
    const int l    = lane & 15;
    const int sel  = lane >> 4;
    const int m0   = mt * 16;
    const int n0   = half * 512 + w * 64;

    const _Float16* ap = xh + (size_t)(m0 + l) * Cv + sel * 16;
    const _Float16* bp[4];
#pragma unroll
    for (int f = 0; f < 4; ++f)
        bp[f] = Wh + (size_t)(n0 + f * 16 + l) * Cv + sel * 16;

    v8f acc[4] = {};
    v16h a_cur = la_load_a(ap);
    v16h b_cur[4];
#pragma unroll
    for (int f = 0; f < 4; ++f) b_cur[f] = *(const v16h*)(bp[f]);

#pragma unroll
    for (int it = 0; it < KSTEPS; ++it) {
        v16h a_nxt;
        v16h b_nxt[4];
        if (it + 1 < KSTEPS) {                 // static under full unroll
            const int cn = (it + 1) * 32;
            a_nxt = la_load_a(ap + cn);
#pragma unroll
            for (int f = 0; f < 4; ++f) b_nxt[f] = *(const v16h*)(bp[f] + cn);
        }
        LA_SCHED_FENCE();   // keep prefetch issued before this step's WMMAs
#pragma unroll
        for (int f = 0; f < 4; ++f)
            acc[f] = WMMA_F32_16x16x32_F16(a_cur, b_cur[f], acc[f]);
        LA_SCHED_FENCE();
        if (it + 1 < KSTEPS) {
            a_cur = a_nxt;
#pragma unroll
            for (int f = 0; f < 4; ++f) b_cur[f] = b_nxt[f];
        }
    }

    const int batch = m0 >> 12;
#pragma unroll
    for (int f = 0; f < 4; ++f) {
        const int col = n0 + f * 16 + l;
        float ks = 0.0f;
#pragma unroll
        for (int r = 0; r < 8; ++r) {
            const int row = m0 + r + sel * 8;
            float v = acc[f][r];
            v = (v > 0.0f) ? (v + 1.0f) : __expf(v);   // elu(v)+1
            if (half == 0) {
                qh[(size_t)row * Cv + col] = (_Float16)v;
            } else {
                kh[(size_t)row * Cv + (col - 512)] = (_Float16)v;
                ks += v;
            }
        }
        if (half == 1) atomicAdd(&ksum[batch * Cv + (col - 512)], ks);
    }
}

// ---------------------------------------------------------------------------
// z[row] = 1 / (dot(q[row,:], ksum[batch,:]) + eps). One wave per row.
// ---------------------------------------------------------------------------
__global__ __launch_bounds__(256)
void la_zfactor_kernel(const _Float16* __restrict__ qh,
                       const float* __restrict__ ksum,
                       float* __restrict__ z) {
    const int w    = threadIdx.x >> 5;
    const int lane = threadIdx.x & 31;
    const int row  = blockIdx.x * 8 + w;
    const int batch = row >> 12;
    const _Float16* qp = qh + (size_t)row * Cv;
    const float*    kp = ksum + batch * Cv;
    float s = 0.0f;
#pragma unroll
    for (int i = 0; i < 16; ++i) {
        const int d = lane + i * 32;
        s += (float)qp[d] * kp[d];
    }
#pragma unroll
    for (int off = 16; off >= 1; off >>= 1) s += __shfl_down(s, off, 32);
    if (lane == 0) z[row] = 1.0f / (s + 1e-6f);
}

// ---------------------------------------------------------------------------
// GEMM2: kv[b,d,e] = sum_n k[b,n,d] * x[b,n,e]  (M=N=512, K=4096 per batch)
// Stage 32(n) x 64 tiles in LDS transposed; next tile's global loads are
// pinned (sched_barrier) ahead of the WMMA compute phase.
// Output stored TRANSPOSED as f16 (kvT[b][e][d]) for GEMM3's B operand.
// ---------------------------------------------------------------------------
__global__ __launch_bounds__(256)
void la_kv_kernel(const _Float16* __restrict__ xh, const _Float16* __restrict__ kh,
                  _Float16* __restrict__ kvT) {
    __shared__ __align__(16) _Float16 kT[64 * 40];   // [d_local][n_local], pad 40
    __shared__ __align__(16) _Float16 xT[64 * 40];   // [e_local][n_local]

    const int b    = blockIdx.y;
    const int dt   = blockIdx.x >> 3;
    const int et   = blockIdx.x & 7;
    const int d0   = dt * 64, e0 = et * 64;
    const int tid  = threadIdx.x;
    const int w    = tid >> 5, lane = tid & 31;
    const int l    = lane & 15, sel = lane >> 4;
    const int wm   = w >> 1, wn = w & 1;
    const int srow = tid >> 3;          // staging: n_local 0..31
    const int scol = (tid & 7) * 8;     // staging: 8 halves per thread

    v8f acc[2] = {};

    size_t grow = ((size_t)b * Nv + srow) * Cv;
    v8h kk = *(const v8h*)(kh + grow + d0 + scol);
    v8h xx = *(const v8h*)(xh + grow + e0 + scol);

#pragma unroll 2
    for (int n0 = 0; n0 < Nv; n0 += 32) {
        __syncthreads();   // previous iteration's LDS reads done
#pragma unroll
        for (int j = 0; j < 8; ++j) {
            kT[(scol + j) * 40 + srow] = kk[j];
            xT[(scol + j) * 40 + srow] = xx[j];
        }
        __syncthreads();

        // prefetch next tile's global data (clamped on last iteration)
        const int n1 = (n0 + 32 < Nv) ? (n0 + 32) : n0;
        grow = ((size_t)b * Nv + n1 + srow) * Cv;
        const v8h kk_n = *(const v8h*)(kh + grow + d0 + scol);
        const v8h xx_n = *(const v8h*)(xh + grow + e0 + scol);
        LA_SCHED_FENCE();   // issue global prefetch before LDS reads + WMMA

        // A fragment: rows are d (k^T), contiguous runs in transposed LDS tile
        const _Float16* ar = &kT[(wm * 16 + l) * 40];
        const v8h alo = *(const v8h*)(ar + sel * 8);
        const v8h ahi = *(const v8h*)(ar + 16 + sel * 8);
        v16h a;
#pragma unroll
        for (int i = 0; i < 8; ++i) { a[i] = alo[i]; a[i + 8] = ahi[i]; }

#pragma unroll
        for (int f = 0; f < 2; ++f) {
            const _Float16* br = &xT[(wn * 32 + f * 16 + l) * 40 + sel * 8];
            const v8h blo = *(const v8h*)(br);
            const v8h bhi = *(const v8h*)(br + 16);
            v16h bfr;
#pragma unroll
            for (int i = 0; i < 8; ++i) { bfr[i] = blo[i]; bfr[i + 8] = bhi[i]; }
            acc[f] = WMMA_F32_16x16x32_F16(a, bfr, acc[f]);
        }
        kk = kk_n;
        xx = xx_n;
    }

    // store D transposed (e-major) as f16
#pragma unroll
    for (int f = 0; f < 2; ++f) {
#pragma unroll
        for (int r = 0; r < 8; ++r) {
            const int d = d0 + wm * 16 + r + sel * 8;
            const int e = e0 + wn * 32 + f * 16 + l;
            kvT[((size_t)b * Cv + e) * Cv + d] = (_Float16)acc[f][r];
        }
    }
}

// ---------------------------------------------------------------------------
// GEMM3: out[row,e] = z[row] * sum_d q[row,d] * kv[b,d,e]
// A = q rows, B = kvT columns (both contiguous). Wave: 16x64, fully unrolled
// software pipeline with sched_barrier-pinned prefetch.
// ---------------------------------------------------------------------------
__global__ __launch_bounds__(256)
void la_out_kernel(const _Float16* __restrict__ qh, const _Float16* __restrict__ kvT,
                   const float* __restrict__ z, float* __restrict__ out) {
    const int mt   = blockIdx.x;        // 0..2047
    const int w    = threadIdx.x >> 5;
    const int lane = threadIdx.x & 31;
    const int l    = lane & 15, sel = lane >> 4;
    const int m0   = mt * 16;
    const int e0   = w * 64;
    const int batch = m0 >> 12;

    const _Float16* ap = qh + (size_t)(m0 + l) * Cv + sel * 16;
    const _Float16* bp[4];
#pragma unroll
    for (int f = 0; f < 4; ++f)
        bp[f] = kvT + (size_t)batch * Cv * Cv + (size_t)(e0 + f * 16 + l) * Cv + sel * 16;

    v8f acc[4] = {};
    v16h a_cur = la_load_a(ap);
    v16h b_cur[4];
#pragma unroll
    for (int f = 0; f < 4; ++f) b_cur[f] = *(const v16h*)(bp[f]);

#pragma unroll
    for (int it = 0; it < KSTEPS; ++it) {
        v16h a_nxt;
        v16h b_nxt[4];
        if (it + 1 < KSTEPS) {                 // static under full unroll
            const int cn = (it + 1) * 32;
            a_nxt = la_load_a(ap + cn);
#pragma unroll
            for (int f = 0; f < 4; ++f) b_nxt[f] = *(const v16h*)(bp[f] + cn);
        }
        LA_SCHED_FENCE();   // keep prefetch issued before this step's WMMAs
#pragma unroll
        for (int f = 0; f < 4; ++f)
            acc[f] = WMMA_F32_16x16x32_F16(a_cur, b_cur[f], acc[f]);
        LA_SCHED_FENCE();
        if (it + 1 < KSTEPS) {
            a_cur = a_nxt;
#pragma unroll
            for (int f = 0; f < 4; ++f) b_cur[f] = b_nxt[f];
        }
    }

#pragma unroll
    for (int f = 0; f < 4; ++f) {
#pragma unroll
        for (int r = 0; r < 8; ++r) {
            const int row = m0 + r + sel * 8;
            const int col = e0 + f * 16 + l;
            out[(size_t)row * Cv + col] = acc[f][r] * z[row];
        }
    }
}

// ---------------------------------------------------------------------------
extern "C" void kernel_launch(void* const* d_in, const int* in_sizes, int n_in,
                              void* d_out, int out_size, void* d_ws, size_t ws_size,
                              hipStream_t stream) {
    const float* x = (const float*)d_in[0];   // [8,4096,512] f32
    const float* W = (const float*)d_in[1];   // [1024,512]   f32
    float* out = (float*)d_out;               // [8,4096,512] f32

    // workspace layout (~101.2 MB)
    char* ws = (char*)d_ws;
    _Float16* xh   = (_Float16*)ws;                        // 32 MB
    _Float16* Wh   = xh  + (size_t)MT * Cv;                //  1 MB
    _Float16* qh   = Wh  + (size_t)2 * Cv * Cv;            // 32 MB
    _Float16* kh   = qh  + (size_t)MT * Cv;                // 32 MB
    _Float16* kvT  = kh  + (size_t)MT * Cv;                //  4 MB
    float*    ksum = (float*)(kvT + (size_t)Bv * Cv * Cv); // 16 KB
    float*    zf   = ksum + Bv * Cv;                       // 128 KB

    const int nx8 = (MT * Cv) / 8;
    const int nw8 = (2 * Cv * Cv) / 8;
    la_cvt_f16_kernel<<<(nx8 + 255) / 256, 256, 0, stream>>>(x, xh, nx8);
    la_cvt_f16_kernel<<<(nw8 + 255) / 256, 256, 0, stream>>>(W, Wh, nw8);
    la_zero_kernel<<<(Bv * Cv + 255) / 256, 256, 0, stream>>>(ksum, Bv * Cv);

    la_qkproj_kernel<<<(MT / 16) * 2, 256, 0, stream>>>(xh, Wh, qh, kh, ksum);
    la_zfactor_kernel<<<MT / 8, 256, 0, stream>>>(qh, ksum, zf);
    la_kv_kernel<<<dim3(64, Bv), 256, 0, stream>>>(xh, kh, kvT);
    la_out_kernel<<<MT / 16, 256, 0, stream>>>(qh, kvT, zf, out);
}
